// MMDLossFunction_76991583748198
// MI455X (gfx1250) — matrix operand order repaired and checked by reading, not compiled, not run on hardware
//
#include <hip/hip_runtime.h>
#include <hip/hip_bf16.h>
#include <stdint.h>

// ---------------------------------------------------------------------------
// MMD loss: xx + yy - 2*xy + penalty, with pair_mean via bf16 WMMA Gram GEMMs
// fused with exp(-d2/2) and global reduction. MI455X / gfx1250, wave32.
// Staging uses GLOBAL_LOAD_ASYNC_TO_LDS_B128 (ASYNCcnt) + double buffering.
// ---------------------------------------------------------------------------

typedef __attribute__((ext_vector_type(16))) __bf16 v16bf;
typedef __attribute__((ext_vector_type(8)))  float  v8f;

#define NROWS 4096
#define KDIM  2500
#define KPAD  2528      // 79 * 32, zero-padded
#define KSTEP 32
#define BM    128
#define BN    128
#define LDSS  40        // LDS row stride in bf16 elems (32 + 8 pad -> 80B rows)

union FragBF { uint4 u[2]; v16bf v; };

__device__ __forceinline__ unsigned short f32_to_bf16_rne(float f) {
    union { float f; unsigned int u; } c; c.f = f;
    unsigned int u = c.u;
    unsigned int lsb = (u >> 16) & 1u;
    u += 0x7fffu + lsb;               // round-to-nearest-even (inputs are finite)
    return (unsigned short)(u >> 16);
}
__device__ __forceinline__ float bf16_to_f32(unsigned short h) {
    union { unsigned int u; float f; } c; c.u = ((unsigned int)h) << 16;
    return c.f;
}

// Async global->LDS 16B copy (per lane). LDS operand = low 32 bits of the
// generic pointer (flat->LDS aperture truncation, ISA 10.2); tracked by ASYNCcnt.
__device__ __forceinline__ void async_b128(unsigned lds_off, unsigned long long gaddr) {
    asm volatile("global_load_async_to_lds_b128 %0, %1, off"
                 :: "v"(lds_off), "v"(gaddr) : "memory");
}
__device__ __forceinline__ void wait_async0() {
    asm volatile("s_wait_asynccnt 0x0" ::: "memory");
}

// Issue async copies for one 128x32 A panel + 128x32 B panel into LDS buffers.
__device__ __forceinline__ void stage_tiles(const unsigned short* __restrict__ Abf,
                                            const unsigned short* __restrict__ Bbf,
                                            unsigned short* AsB, unsigned short* BsB,
                                            int rowBase, int colBase, int k0, int tid) {
    #pragma unroll
    for (int q = 0; q < 2; ++q) {
        const int chunk = tid + q * 256;     // 0..511 -> 128 rows x 4 chunks of 16B
        const int r   = chunk >> 2;
        const int c16 = chunk & 3;
        async_b128((unsigned)(uintptr_t)(AsB + r * LDSS + c16 * 8),
                   (unsigned long long)(uintptr_t)(Abf + (size_t)(rowBase + r) * KPAD + k0 + c16 * 8));
        async_b128((unsigned)(uintptr_t)(BsB + r * LDSS + c16 * 8),
                   (unsigned long long)(uintptr_t)(Bbf + (size_t)(colBase + r) * KPAD + k0 + c16 * 8));
    }
}

// --- Kernel 1: f32 -> bf16 (padded) + row squared-norms from the bf16 data ---
__global__ __launch_bounds__(256)
void convert_rows(const float* __restrict__ x, const float* __restrict__ y,
                  unsigned short* __restrict__ xb, unsigned short* __restrict__ yb,
                  float* __restrict__ x2, float* __restrict__ y2) {
    const int row = blockIdx.x;
    const float*    src = (blockIdx.y == 0) ? x  : y;
    unsigned short* dst = (blockIdx.y == 0) ? xb : yb;
    float*          nrm = (blockIdx.y == 0) ? x2 : y2;

    float acc = 0.f;
    for (int c = threadIdx.x; c < KPAD; c += blockDim.x) {
        float v = (c < KDIM) ? src[(size_t)row * KDIM + c] : 0.f;
        unsigned short h = f32_to_bf16_rne(v);
        dst[(size_t)row * KPAD + c] = h;
        float g = bf16_to_f32(h);
        acc += g * g;
    }
    __shared__ float red[8];
    for (int off = 16; off > 0; off >>= 1) acc += __shfl_down(acc, off, 32);
    const int lane = threadIdx.x & 31, wave = threadIdx.x >> 5;
    if (lane == 0) red[wave] = acc;
    __syncthreads();
    if (threadIdx.x == 0) {
        float s = 0.f;
        #pragma unroll
        for (int w = 0; w < 8; ++w) s += red[w];
        nrm[row] = s;
    }
}

// --- Kernel 2: zero the three accumulators ---
__global__ void zero_acc(float* acc) {
    if (threadIdx.x < 3) acc[threadIdx.x] = 0.f;
}

// --- Kernel 3: fused Gram GEMM (bf16 WMMA) + exp(-d2/2) + reduction ---
__global__ __launch_bounds__(256)
void gram_exp_reduce(const unsigned short* __restrict__ Abf,
                     const unsigned short* __restrict__ Bbf,
                     const float* __restrict__ a2, const float* __restrict__ b2,
                     float* __restrict__ accum) {
    __shared__ unsigned short As[2][BM * LDSS];   // 2 x 10,240 B (double buffer)
    __shared__ unsigned short Bs[2][BN * LDSS];
    __shared__ float a2s[BM];
    __shared__ float b2s[BN];
    __shared__ float red[8];

    const int tid  = threadIdx.x;     // 0..255, 8 waves
    const int wave = tid >> 5;
    const int lane = tid & 31;
    const int wm   = wave >> 2;       // 0..1 -> 64-row slab
    const int wn   = wave & 3;        // 0..3 -> 32-col slab
    const int rowBase = blockIdx.y * BM;
    const int colBase = blockIdx.x * BN;
    const int halfsel = lane >> 4;    // upper/lower 16 lanes
    const int lr      = lane & 15;

    if (tid < BM) a2s[tid]      = a2[rowBase + tid];
    else          b2s[tid - BM] = b2[colBase + (tid - BM)];

    v8f c[4][2];
    const v8f vzero = {0.f, 0.f, 0.f, 0.f, 0.f, 0.f, 0.f, 0.f};
    #pragma unroll
    for (int i = 0; i < 4; ++i)
        #pragma unroll
        for (int j = 0; j < 2; ++j) c[i][j] = vzero;

    const int NSTEPS = KPAD / KSTEP;  // 79
    // Prologue: issue async staging for step 0 into buffer 0.
    stage_tiles(Abf, Bbf, &As[0][0], &Bs[0][0], rowBase, colBase, 0, tid);

    for (int step = 0; step < NSTEPS; ++step) {
        wait_async0();                 // this wave's async writes are in LDS
        __syncthreads();               // ... and so are every other wave's
        const int cur = step & 1;
        if (step + 1 < NSTEPS)         // prefetch next panel into other buffer
            stage_tiles(Abf, Bbf, &As[cur ^ 1][0], &Bs[cur ^ 1][0],
                        rowBase, colBase, (step + 1) * KSTEP, tid);

        const unsigned short* Ac = &As[cur][0];
        const unsigned short* Bc = &Bs[cur][0];

        // A fragments (ISA 16-bit A 16x32 layout): per lane K = {h*8+0..7, 16+h*8+0..7}
        FragBF af[4];
        #pragma unroll
        for (int i = 0; i < 4; ++i) {
            const unsigned short* p = Ac + (wm * 64 + i * 16 + lr) * LDSS + halfsel * 8;
            af[i].u[0] = *(const uint4*)(p);
            af[i].u[1] = *(const uint4*)(p + 16);
        }
        // B fragments (row of B striped across lanes): per lane K = {h*8+0..7, 16+h*8+0..7}
        FragBF bfr[2];
        #pragma unroll
        for (int j = 0; j < 2; ++j) {
            const unsigned short* p = Bc + (wn * 32 + j * 16 + lr) * LDSS + halfsel * 8;
            bfr[j].u[0] = *(const uint4*)(p);
            bfr[j].u[1] = *(const uint4*)(p + 16);
        }
        #pragma unroll
        for (int i = 0; i < 4; ++i)
            #pragma unroll
            for (int j = 0; j < 2; ++j)
                c[i][j] = __builtin_amdgcn_wmma_f32_16x16x32_bf16(
                    false, af[i].v, false, bfr[j].v, (short)0, c[i][j], false, false);
    }

    // Epilogue: C layout -> lane holds (M = v + 8*halfsel, N = lr) per VGPR v.
    float s = 0.f;
    #pragma unroll
    for (int i = 0; i < 4; ++i) {
        #pragma unroll
        for (int j = 0; j < 2; ++j) {
            #pragma unroll
            for (int v = 0; v < 8; ++v) {
                const int lm = wm * 64 + i * 16 + v + 8 * halfsel;
                const int ln = wn * 32 + j * 16 + lr;
                float d2 = a2s[lm] + b2s[ln] - 2.0f * c[i][j][v];
                d2 = fmaxf(d2, 0.f);
                s += __expf(-0.5f * d2);
            }
        }
    }
    for (int off = 16; off > 0; off >>= 1) s += __shfl_down(s, off, 32);
    if (lane == 0) red[wave] = s;
    __syncthreads();
    if (tid == 0) {
        float t = 0.f;
        #pragma unroll
        for (int w = 0; w < 8; ++w) t += red[w];
        atomicAdd(accum, t);
    }
}

// --- Kernel 4: combine sums + penalty ---
__global__ void mmd_finalize(const float* __restrict__ acc,
                             const float* __restrict__ avg_step,
                             float* __restrict__ out) {
    if (threadIdx.x == 0 && blockIdx.x == 0) {
        const float inv = 1.0f / ((float)NROWS * (float)NROWS);
        const float xx = acc[0] * inv;
        const float xy = acc[1] * inv;
        const float yy = acc[2] * inv;
        const float pen = (fmaxf(1.0f, avg_step[0]) - 1.0f) * 0.002f;
        out[0] = xx + yy - 2.0f * xy + pen;
    }
}

extern "C" void kernel_launch(void* const* d_in, const int* in_sizes, int n_in,
                              void* d_out, int out_size, void* d_ws, size_t ws_size,
                              hipStream_t stream) {
    const float* x        = (const float*)d_in[0];
    const float* y        = (const float*)d_in[1];
    const float* avg_step = (const float*)d_in[2];

    char* ws = (char*)d_ws;
    const size_t szb = (size_t)NROWS * KPAD * sizeof(unsigned short); // 20,709,376 B
    unsigned short* xb = (unsigned short*)(ws);
    unsigned short* yb = (unsigned short*)(ws + szb);
    float* x2  = (float*)(ws + 2 * szb);
    float* y2  = x2 + NROWS;
    float* acc = y2 + NROWS;   // 3 floats: xx, xy, yy sums

    convert_rows<<<dim3(NROWS, 2), 256, 0, stream>>>(x, y, xb, yb, x2, y2);
    zero_acc<<<1, 32, 0, stream>>>(acc);

    const dim3 ggrid(NROWS / BN, NROWS / BM);   // 32 x 32 blocks
    gram_exp_reduce<<<ggrid, 256, 0, stream>>>(xb, xb, x2, x2, acc + 0);
    gram_exp_reduce<<<ggrid, 256, 0, stream>>>(xb, yb, x2, y2, acc + 1);
    gram_exp_reduce<<<ggrid, 256, 0, stream>>>(yb, yb, y2, y2, acc + 2);

    mmd_finalize<<<1, 1, 0, stream>>>(acc, avg_step, (float*)d_out);
}
